// CustomLSTM_15401752724068
// MI455X (gfx1250) — compile-verified, matrix operand order checked by reading, not run on hardware
//
#include <hip/hip_runtime.h>

#define B_  64
#define S_  512
#define I_  1024
#define H_  1024
#define G4H 4096
#define BPAD 40   // padded LDS row stride (elements) for staged B panel

typedef __attribute__((ext_vector_type(16))) __bf16 v16bf;
typedef __attribute__((ext_vector_type(8)))  __bf16 v8bf;
typedef __attribute__((ext_vector_type(8)))  float  v8f;

// ---------- bf16 helpers (round-to-nearest-even) ----------
__device__ __forceinline__ __bf16 f2bf(float f) {
  union { float f; unsigned u; } x; x.f = f;
  unsigned r = x.u + 0x7FFFu + ((x.u >> 16) & 1u);
  unsigned short h = (unsigned short)(r >> 16);
  union { unsigned short s; __bf16 b; } y; y.s = h;
  return y.b;
}
__device__ __forceinline__ float bf2f(__bf16 b) {
  union { unsigned short s; __bf16 b; } x; x.b = b;
  union { unsigned u; float f; } y; y.u = ((unsigned)x.s) << 16;
  return y.f;
}

// 16x32 bf16 WMMA fragment from global (A layout; B identical on the
// pre-transposed weights): lane L holds row (L&15); elements 0..7 are
// K = half*8+0..7 (one b128), elements 8..15 are K = 16+half*8+0..7.
__device__ __forceinline__ v16bf load_frag(const __bf16* __restrict__ base,
                                           int stride, int row0, int k0, int lane) {
  int m = lane & 15, half = lane >> 4;
  const __bf16* p = base + (size_t)(row0 + m) * stride + k0 + half * 8;
  v8bf lo = *reinterpret_cast<const v8bf*>(p);
  v8bf hi = *reinterpret_cast<const v8bf*>(p + 16);
  return __builtin_shufflevector(lo, hi, 0,1,2,3,4,5,6,7,8,9,10,11,12,13,14,15);
}

// Same fragment pattern out of a staged LDS panel laid out [n][BPAD].
__device__ __forceinline__ v16bf lds_frag(const __bf16* base, int n0, int lane) {
  int n = lane & 15, half = lane >> 4;
  const __bf16* p = base + (n0 + n) * BPAD + half * 8;
  v8bf lo = *reinterpret_cast<const v8bf*>(p);        // ds_load_b128
  v8bf hi = *reinterpret_cast<const v8bf*>(p + 16);   // ds_load_b128
  return __builtin_shufflevector(lo, hi, 0,1,2,3,4,5,6,7,8,9,10,11,12,13,14,15);
}

// ---------- gfx1250 async global->LDS copy (ASYNCcnt-tracked) ----------
__device__ __forceinline__ void async_b128(void* lds_ptr, const __bf16* g) {
  unsigned loff = (unsigned)(size_t)lds_ptr;   // generic LDS addr: low 32 bits
  asm volatile("global_load_async_to_lds_b128 %0, %1, off"
               :: "v"(loff), "v"((unsigned long long)(size_t)g)
               : "memory");
}
__device__ __forceinline__ void wait_async0() {
#if __has_builtin(__builtin_amdgcn_s_wait_asynccnt)
  __builtin_amdgcn_s_wait_asynccnt(0);
#else
  asm volatile("s_wait_asynccnt 0x0" ::: "memory");
#endif
}

#define WMMA_BF16(A, Bm, C) \
  __builtin_amdgcn_wmma_f32_16x16x32_bf16(false, (A), false, (Bm), (short)0, (C), false, false)

// ---------- one-time conversion kernels ----------
__global__ void cvt_x(const float* __restrict__ x, __bf16* __restrict__ xb, int n) {
  int i = blockIdx.x * blockDim.x + threadIdx.x;
  if (i < n) xb[i] = f2bf(x[i]);
}

// dst[(g*H + n)*I + k] = src_g[k*H + n]  (transposed, concatenated, bf16)
__global__ void cvt_T(const float* __restrict__ m0, const float* __restrict__ m1,
                      const float* __restrict__ m2, const float* __restrict__ m3,
                      __bf16* __restrict__ dst) {
  int idx = blockIdx.x * blockDim.x + threadIdx.x;
  int k  = idx & (I_ - 1);
  int ng = idx >> 10;
  int g  = ng >> 10;
  int nl = ng & (H_ - 1);
  const float* src = (g == 0) ? m0 : (g == 1) ? m1 : (g == 2) ? m2 : m3;
  dst[idx] = f2bf(src[(size_t)k * H_ + nl]);
}

__global__ void zero_state(__bf16* hA, __bf16* hB, float* c) {
  int i = blockIdx.x * blockDim.x + threadIdx.x;
  if (i < B_ * H_) { hA[i] = f2bf(0.f); hB[i] = f2bf(0.f); c[i] = 0.f; }
}

// ---------- Phase 1: x_proj = xb @ U_T^T + bias  (M=32768, N=4096, K=1024) ----------
// Block = 256 threads = 8 waves. Block tile 128x64. The 64x32 B K-panel is
// shared by all 8 waves: double-buffered in LDS, filled by async copies that
// overlap the WMMAs; A fragments are register-pipelined from global.
__global__ void __launch_bounds__(256)
lstm_gemm1(const __bf16* __restrict__ xb, const __bf16* __restrict__ UT,
           const float* __restrict__ bfp, const float* __restrict__ bip,
           const float* __restrict__ bcp, const float* __restrict__ bop,
           __bf16* __restrict__ xp) {
  __shared__ __align__(16) __bf16 Bsh[2][64 * BPAD];
  int tid  = threadIdx.x;
  int lane = tid & 31;
  int wave = tid >> 5;
  int row0  = (blockIdx.y * 8 + wave) * 16;
  int nbase = blockIdx.x * 64;
  const float* bias = (nbase >> 10) == 0 ? bfp :
                      (nbase >> 10) == 1 ? bip :
                      (nbase >> 10) == 2 ? bcp : bop;

  // async panel-fill mapping: thread -> (row 0..63, 16B chunk 0..3)
  int prow = tid >> 2;
  int pchk = (tid & 3) * 8;                       // element offset in K
  const __bf16* gB = UT + (size_t)(nbase + prow) * I_ + pchk;

  // prologue: panel k0=0 into buffer 0, first A fragment into regs
  async_b128(&Bsh[0][prow * BPAD + pchk], gB);
  v16bf Ac = load_frag(xb, I_, row0, 0, lane);
  wait_async0();
  __syncthreads();

  v8f a0 = {}, a1 = {}, a2 = {}, a3 = {};
  int buf = 0;
  for (int k0 = 0; k0 < I_; k0 += 32) {
    bool notlast = (k0 + 32) < I_;
    if (notlast)                                   // fill other buffer in the shadow
      async_b128(&Bsh[buf ^ 1][prow * BPAD + pchk], gB + (k0 + 32));
    v16bf An = Ac;
    if (notlast) An = load_frag(xb, I_, row0, k0 + 32, lane);

    const __bf16* pb = &Bsh[buf][0];
    v16bf B0 = lds_frag(pb,  0, lane);
    v16bf B1 = lds_frag(pb, 16, lane);
    v16bf B2 = lds_frag(pb, 32, lane);
    v16bf B3 = lds_frag(pb, 48, lane);
    a0 = WMMA_BF16(Ac, B0, a0);
    a1 = WMMA_BF16(Ac, B1, a1);
    a2 = WMMA_BF16(Ac, B2, a2);
    a3 = WMMA_BF16(Ac, B3, a3);

    if (notlast) wait_async0();
    __syncthreads();
    Ac = An;
    buf ^= 1;
  }

  int n = lane & 15, half = lane >> 4;
  for (int r = 0; r < 8; ++r) {
    int row = row0 + r + 8 * half;
    int c0  = nbase + n;
    size_t base = (size_t)row * G4H + c0;
    xp[base     ] = f2bf(a0[r] + bias[(c0     ) & 1023]);
    xp[base + 16] = f2bf(a1[r] + bias[(c0 + 16) & 1023]);
    xp[base + 32] = f2bf(a2[r] + bias[(c0 + 32) & 1023]);
    xp[base + 48] = f2bf(a3[r] + bias[(c0 + 48) & 1023]);
  }
}

// ---------- Phase 2: one recurrent step ----------
// Block = 128 threads = 4 waves; wave g owns gate g's 16x16 tile. K-loop is
// software-pipelined in registers (distinct next-fragment set) so the 32
// sequential WMMAs overlap their feeding loads. Gate fusion through LDS.
__global__ void __launch_bounds__(128)
lstm_step(const __bf16* __restrict__ hin, const __bf16* __restrict__ WT,
          const __bf16* __restrict__ xp, float* __restrict__ cbuf,
          __bf16* __restrict__ hout, float* __restrict__ out, int t) {
  __shared__ float gsh[4 * 256];
  int lane = threadIdx.x & 31;
  int g    = threadIdx.x >> 5;
  int row0 = blockIdx.x * 16;
  int j0   = blockIdx.y * 16;
  int nrow = g * H_ + j0;

  v8f acc = {};
  v16bf Ac = load_frag(hin, H_, row0, 0, lane);
  v16bf Bc = load_frag(WT,  H_, nrow, 0, lane);
  for (int k0 = 0; k0 < H_ - 32; k0 += 32) {
    v16bf An = load_frag(hin, H_, row0, k0 + 32, lane);
    v16bf Bn = load_frag(WT,  H_, nrow, k0 + 32, lane);
    acc = WMMA_BF16(Ac, Bc, acc);
    Ac = An;
    Bc = Bn;
  }
  acc = WMMA_BF16(Ac, Bc, acc);

  int n = lane & 15, half = lane >> 4;
  for (int r = 0; r < 8; ++r) {
    int m = r + 8 * half;
    int b = row0 + m;
    int col = g * H_ + j0 + n;
    gsh[g * 256 + m * 16 + n] = acc[r] + bf2f(xp[((size_t)b * S_ + t) * G4H + col]);
  }
  __syncthreads();
  for (int e = threadIdx.x; e < 256; e += 128) {
    int m = e >> 4, n2 = e & 15;
    int b = row0 + m, j = j0 + n2;
    float gf = gsh[0 * 256 + e];
    float gi = gsh[1 * 256 + e];
    float gc = gsh[2 * 256 + e];
    float go = gsh[3 * 256 + e];
    float f  = 1.f / (1.f + __expf(-gf));
    float i  = 1.f / (1.f + __expf(-gi));
    float cb = tanhf(gc);
    float o  = 1.f / (1.f + __expf(-go));
    float cn = f * cbuf[b * H_ + j] + i * cb;
    float hn = o * tanhf(cn);
    cbuf[b * H_ + j] = cn;
    hout[b * H_ + j] = f2bf(hn);
    out[(size_t)2 * B_ * H_ + ((size_t)b * S_ + t) * H_ + j] = hn;  // hidden_seq
    if (t == S_ - 1) {
      out[b * H_ + j]           = hn;  // h_T
      out[B_ * H_ + b * H_ + j] = cn;  // c_T
    }
  }
}

extern "C" void kernel_launch(void* const* d_in, const int* in_sizes, int n_in,
                              void* d_out, int out_size, void* d_ws, size_t ws_size,
                              hipStream_t stream) {
  const float* x  = (const float*)d_in[0];
  const float* Uf = (const float*)d_in[1];
  const float* Ui = (const float*)d_in[2];
  const float* Uc = (const float*)d_in[3];
  const float* Uo = (const float*)d_in[4];
  const float* Wf = (const float*)d_in[5];
  const float* Wi = (const float*)d_in[6];
  const float* Wc = (const float*)d_in[7];
  const float* Wo = (const float*)d_in[8];
  const float* bf = (const float*)d_in[9];
  const float* bi = (const float*)d_in[10];
  const float* bc = (const float*)d_in[11];
  const float* bo = (const float*)d_in[12];
  float* out = (float*)d_out;

  char* ws = (char*)d_ws;
  size_t off = 0;
  __bf16* xb = (__bf16*)(ws + off); off += (size_t)B_ * S_ * I_ * 2;   // 64 MB
  __bf16* UT = (__bf16*)(ws + off); off += (size_t)G4H * I_ * 2;       // 8 MB
  __bf16* WT = (__bf16*)(ws + off); off += (size_t)G4H * H_ * 2;       // 8 MB
  __bf16* xp = (__bf16*)(ws + off); off += (size_t)B_ * S_ * G4H * 2;  // 256 MB
  __bf16* hA = (__bf16*)(ws + off); off += (size_t)B_ * H_ * 2;
  __bf16* hB = (__bf16*)(ws + off); off += (size_t)B_ * H_ * 2;
  float*  cb2 = (float*)(ws + off); off += (size_t)B_ * H_ * 4;

  cvt_x<<<(B_ * S_ * I_ + 255) / 256, 256, 0, stream>>>(x, xb, B_ * S_ * I_);
  cvt_T<<<(G4H * I_) / 256, 256, 0, stream>>>(Uf, Ui, Uc, Uo, UT);
  cvt_T<<<(G4H * H_) / 256, 256, 0, stream>>>(Wf, Wi, Wc, Wo, WT);
  zero_state<<<(B_ * H_ + 255) / 256, 256, 0, stream>>>(hA, hB, cb2);

  lstm_gemm1<<<dim3(64, 256), 256, 0, stream>>>(xb, UT, bf, bi, bc, bo, xp);

  for (int t = 0; t < S_; ++t) {
    const __bf16* hin = (t & 1) ? hB : hA;
    __bf16*      hou = (t & 1) ? hA : hB;
    lstm_step<<<dim3(B_ / 16, H_ / 16), 128, 0, stream>>>(hin, WT, xp, cb2, hou, out, t);
  }
}
